// PowerSoftmax_61289183314190
// MI455X (gfx1250) — matrix operand order repaired and checked by reading, not compiled, not run on hardware
//
#include <hip/hip_runtime.h>
#include <stdint.h>

// Power-softmax attention (P=2), MI455X / gfx1250, wave32 WMMA f16->f32,
// with async global->LDS double-buffered K/V pipeline.
// out = (sum_m s_m^2 * v_m) / (sum_m s_m^2 + eps),  s = (q . k) / 8

#define BDIM   4
#define HDIM   16
#define NSEQ   2048
#define DHEAD  64
#define BM     64        // query rows per workgroup (16 per wave)
#define BN     64        // keys per main-loop iteration
#define NKB    (NSEQ / BN)
#define NWAVES 4
#define LSTR   72        // LDS row stride in halves (64 data + 8 pad)
#define SCALE  0.125f    // 1/sqrt(64)
#define EPSF   1e-6f

typedef _Float16 v16h __attribute__((ext_vector_type(16)));
typedef _Float16 v8h  __attribute__((ext_vector_type(8)));
typedef _Float16 v4h  __attribute__((ext_vector_type(4)));
typedef float    v8f  __attribute__((ext_vector_type(8)));

// Low 32 bits of a generic pointer to __shared__ = wave-relative LDS byte
// address (addrspacecast(3->0) puts the LDS offset in the low dword).
__device__ __forceinline__ unsigned lds_addr32(const void* p) {
  return (unsigned)(uintptr_t)p;
}

// Load halves p[0..7] and p[16..23] (16B-aligned) as two v8h and concat.
__device__ __forceinline__ v16h ld16h(const _Float16* p) {
  v8h lo = *(const v8h*)p;
  v8h hh = *(const v8h*)(p + 16);
  return __builtin_shufflevector(lo, hh, 0,1,2,3,4,5,6,7,8,9,10,11,12,13,14,15);
}

// ---------------------------------------------------------------------------
// Pre-pass: K -> row-major f16 Kh[bh][key][d];  V -> transposed f16
// Vt[bh][d][key].  LDS-tiled so both global reads and writes are coalesced.
// ---------------------------------------------------------------------------
__global__ __launch_bounds__(128)
void convert_kv_kernel(const float* __restrict__ k, const float* __restrict__ v,
                       _Float16* __restrict__ kh, _Float16* __restrict__ vt) {
  __shared__ _Float16 Kt[BN * LSTR];
  __shared__ _Float16 Vtt[DHEAD * LSTR];
  const int tid  = threadIdx.x;
  const int bh   = blockIdx.y;
  const int key0 = blockIdx.x * BN;
  const float* K = k + (size_t)bh * NSEQ * DHEAD;
  const float* V = v + (size_t)bh * NSEQ * DHEAD;

  #pragma unroll
  for (int it = 0; it < 8; ++it) {
    int idx = it * 128 + tid;
    int r   = idx >> 4;
    int c4  = (idx & 15) << 2;
    float4 kk4 = *(const float4*)(K + (size_t)(key0 + r) * DHEAD + c4);
    float4 vv4 = *(const float4*)(V + (size_t)(key0 + r) * DHEAD + c4);
    v4h ks;
    ks[0] = (_Float16)kk4.x; ks[1] = (_Float16)kk4.y;
    ks[2] = (_Float16)kk4.z; ks[3] = (_Float16)kk4.w;
    *(v4h*)&Kt[r * LSTR + c4] = ks;
    Vtt[(c4 + 0) * LSTR + r] = (_Float16)vv4.x;
    Vtt[(c4 + 1) * LSTR + r] = (_Float16)vv4.y;
    Vtt[(c4 + 2) * LSTR + r] = (_Float16)vv4.z;
    Vtt[(c4 + 3) * LSTR + r] = (_Float16)vv4.w;
  }
  __syncthreads();

  _Float16* khb = kh + (size_t)bh * NSEQ * DHEAD;
  _Float16* vtb = vt + (size_t)bh * DHEAD * NSEQ;
  #pragma unroll
  for (int it = 0; it < 4; ++it) {
    int idx = it * 128 + tid;
    int r   = idx >> 3;          // key row (K) or column (V)
    int ch  = (idx & 7) * 8;     // 8-half chunk
    v8h kk = *(const v8h*)&Kt[r * LSTR + ch];
    v8h vv = *(const v8h*)&Vtt[r * LSTR + ch];
    *(v8h*)(khb + (size_t)(key0 + r) * DHEAD + ch) = kk;
    *(v8h*)(vtb + (size_t)r * NSEQ + key0 + ch)    = vv;
  }
}

// ---------------------------------------------------------------------------
// Main kernel: async double-buffered K/V tiles, WMMA f16->f32.
// ---------------------------------------------------------------------------
__global__ __launch_bounds__(128)
void power_attn_p2_async_kernel(const float* __restrict__ q,
                                const _Float16* __restrict__ kh,
                                const _Float16* __restrict__ vt,
                                float* __restrict__ out) {
  __shared__ _Float16 Ks[2][BN * LSTR];     // [buf][key][d]      f16
  __shared__ _Float16 Vs[2][DHEAD * LSTR];  // [buf][d][key]      f16 (transposed)
  __shared__ _Float16 Wb[NWAVES * 16 * LSTR];

  const int tid  = threadIdx.x;
  const int lane = tid & 31;
  const int wave = tid >> 5;
  const int l15  = lane & 15;
  const int hi   = lane >> 4;

  const int bh   = blockIdx.y;
  const int row0 = blockIdx.x * BM + wave * 16;

  const float*    Q  = q  + (size_t)bh * NSEQ * DHEAD;
  const _Float16* Kg = kh + (size_t)bh * NSEQ * DHEAD;
  const _Float16* Vg = vt + (size_t)bh * DHEAD * NSEQ;
  float*          O  = out + (size_t)bh * NSEQ * DHEAD;

  const unsigned ksb[2] = { lds_addr32(&Ks[0][0]), lds_addr32(&Ks[1][0]) };
  const unsigned vsb[2] = { lds_addr32(&Vs[0][0]), lds_addr32(&Vs[1][0]) };

  // Async fill of one 64-key K/V tile pair into buffer `buf`.
  // 128 threads x 4 iters x 16B cover 64 rows x 128B per matrix.
  auto issue_async = [&](int buf, int key0) {
    #pragma unroll
    for (int it = 0; it < 4; ++it) {
      int idx = it * 128 + tid;
      int r   = idx >> 3;           // K: key row; V: column
      int ch  = idx & 7;            // 16B chunk within the 128B row
      unsigned lk = ksb[buf] + (unsigned)(r * (LSTR * 2) + ch * 16);
      unsigned lv = vsb[buf] + (unsigned)(r * (LSTR * 2) + ch * 16);
      const _Float16* gk = Kg + (size_t)(key0 + r) * DHEAD + ch * 8;
      const _Float16* gv = Vg + (size_t)r * NSEQ + key0 + ch * 8;
      asm volatile("global_load_async_to_lds_b128 %0, %1, off"
                   :: "v"(lk), "v"((unsigned long long)(uintptr_t)gk) : "memory");
      asm volatile("global_load_async_to_lds_b128 %0, %1, off"
                   :: "v"(lv), "v"((unsigned long long)(uintptr_t)gv) : "memory");
    }
  };

  // Kick off the first tile before doing anything else.
  issue_async(0, 0);

  // ---- Q tile (16x64) -> A-matrix layout regs, scaled by 1/8, f32->f16 ----
  // A layout (16x32 f16): lane l15 = M row; e[i] = K = 8*hi+i (i<8),
  //                                         e[8+i] = K = 16+8*hi+i.
  v16h qa[2];
  {
    const float* qrow = Q + (size_t)(row0 + l15) * DHEAD;
    #pragma unroll
    for (int dc = 0; dc < 2; ++dc) {
      const float4* p0 = (const float4*)(qrow + dc * 32 + 8 * hi);
      const float4* p1 = (const float4*)(qrow + dc * 32 + 16 + 8 * hi);
      float4 a0 = p0[0], a1 = p0[1];
      float4 b0 = p1[0], b1 = p1[1];
      qa[dc][0]  = (_Float16)(a0.x * SCALE); qa[dc][1]  = (_Float16)(a0.y * SCALE);
      qa[dc][2]  = (_Float16)(a0.z * SCALE); qa[dc][3]  = (_Float16)(a0.w * SCALE);
      qa[dc][4]  = (_Float16)(a1.x * SCALE); qa[dc][5]  = (_Float16)(a1.y * SCALE);
      qa[dc][6]  = (_Float16)(a1.z * SCALE); qa[dc][7]  = (_Float16)(a1.w * SCALE);
      qa[dc][8]  = (_Float16)(b0.x * SCALE); qa[dc][9]  = (_Float16)(b0.y * SCALE);
      qa[dc][10] = (_Float16)(b0.z * SCALE); qa[dc][11] = (_Float16)(b0.w * SCALE);
      qa[dc][12] = (_Float16)(b1.x * SCALE); qa[dc][13] = (_Float16)(b1.y * SCALE);
      qa[dc][14] = (_Float16)(b1.z * SCALE); qa[dc][15] = (_Float16)(b1.w * SCALE);
    }
  }

  v8f   o[4] = {};
  float rs[8] = {0.f,0.f,0.f,0.f,0.f,0.f,0.f,0.f};
  _Float16* wrow = &Wb[wave * 16 * LSTR];

  for (int kb = 0; kb < NKB; ++kb) {
    const int cur = kb & 1;

    // Wait this wave's async loads, then publish tile `cur` to all waves.
    // The same barrier proves every wave finished reading tile cur^1 last
    // iteration, so issuing into cur^1 below is WAR-safe.
    asm volatile("s_wait_asynccnt 0x0" ::: "memory");
    __syncthreads();

    if (kb + 1 < NKB) issue_async(cur ^ 1, (kb + 1) * BN);

    const _Float16* KsB = &Ks[cur][0];
    const _Float16* VsB = &Vs[cur][0];

    // ---- S = Qs @ K^T for four 16-key tiles; w = s^2 stashed to Wb ----
    #pragma unroll
    for (int n0t = 0; n0t < 4; ++n0t) {
      const int n0 = n0t * 16;
      v8f s = {};
      #pragma unroll
      for (int dc = 0; dc < 2; ++dc) {
        // B layout (32x16 f16): lane l15 = N col; e[i] = K[n0+l15][dc*32+8*hi+i], e[8+i] = K[n0+l15][dc*32+16+8*hi+i]
        v16h bk = ld16h(&KsB[(n0 + l15) * LSTR + dc * 32 + 8 * hi]);
        s = __builtin_amdgcn_wmma_f32_16x16x32_f16(false, qa[dc], false, bk,
                                                   (short)0, s, false, false);
      }
      // C layout: reg r <-> row (r + 8*hi), col l15 (within this key tile)
      #pragma unroll
      for (int r = 0; r < 8; ++r) {
        float e = s[r];
        float w = e * e;
        rs[r] += w;
        wrow[(r + 8 * hi) * LSTR + n0 + l15] = (_Float16)w;
      }
    }
    // Same-wave DS ops are in-order: wait for the w stores, no barrier needed.
    asm volatile("s_wait_dscnt 0x0" ::: "memory");

    // ---- O += W @ V over two 32-key chunks ----
    #pragma unroll
    for (int kc = 0; kc < 2; ++kc) {
      const _Float16* wp = &wrow[l15 * LSTR + kc * 32 + 8 * hi];
      v8h wlo = *(const v8h*)wp;
      v8h whi = *(const v8h*)(wp + 16);
      v16h aw = __builtin_shufflevector(wlo, whi,
                                        0,1,2,3,4,5,6,7,8,9,10,11,12,13,14,15);
      #pragma unroll
      for (int nt = 0; nt < 4; ++nt) {
        // B: e[i] = V[kc*32 + 8*hi + i][nt*16 + l15], e[8+i] = V[kc*32 + 16 + 8*hi + i][nt*16 + l15]
        v16h bv = ld16h(&VsB[(nt * 16 + l15) * LSTR + kc * 32 + 8 * hi]);
        o[nt] = __builtin_amdgcn_wmma_f32_16x16x32_f16(false, aw, false, bv,
                                                       (short)0, o[nt], false, false);
      }
    }
  }

  // ---- normalize and store ----
  #pragma unroll
  for (int r = 0; r < 8; ++r) {
    float t = rs[r];
    t += __shfl_xor(t, 1, 32);   // masks <= 8 stay inside each 16-lane half,
    t += __shfl_xor(t, 2, 32);   // matching this reg's row = r + 8*hi
    t += __shfl_xor(t, 4, 32);
    t += __shfl_xor(t, 8, 32);
    const float inv = 1.0f / (t + EPSF);
    const int orow = row0 + r + 8 * hi;
    #pragma unroll
    for (int nt = 0; nt < 4; ++nt) {
      O[(size_t)orow * DHEAD + nt * 16 + l15] = o[nt][r] * inv;
    }
  }
}

// ---------------------------------------------------------------------------
// Fallback (round-1 kernel): in-loop conversion, no workspace needed.
// ---------------------------------------------------------------------------
__global__ __launch_bounds__(128)
void power_attn_p2_fallback(const float* __restrict__ q,
                            const float* __restrict__ k,
                            const float* __restrict__ v,
                            float* __restrict__ out) {
  __shared__ _Float16 Ks[BN * LSTR];
  __shared__ _Float16 Vt[DHEAD * LSTR];
  __shared__ _Float16 Wb[NWAVES * 16 * LSTR];

  const int tid  = threadIdx.x;
  const int lane = tid & 31;
  const int wave = tid >> 5;
  const int l15  = lane & 15;
  const int hi   = lane >> 4;

  const int bh   = blockIdx.y;
  const int row0 = blockIdx.x * BM + wave * 16;

  const float* Q = q + (size_t)bh * NSEQ * DHEAD;
  const float* K = k + (size_t)bh * NSEQ * DHEAD;
  const float* V = v + (size_t)bh * NSEQ * DHEAD;
  float*       O = out + (size_t)bh * NSEQ * DHEAD;

  v16h qa[2];
  {
    const float* qrow = Q + (size_t)(row0 + l15) * DHEAD;
    #pragma unroll
    for (int dc = 0; dc < 2; ++dc) {
      const float4* p0 = (const float4*)(qrow + dc * 32 + 8 * hi);
      const float4* p1 = (const float4*)(qrow + dc * 32 + 16 + 8 * hi);
      float4 a0 = p0[0], a1 = p0[1];
      float4 b0 = p1[0], b1 = p1[1];
      qa[dc][0]  = (_Float16)(a0.x * SCALE); qa[dc][1]  = (_Float16)(a0.y * SCALE);
      qa[dc][2]  = (_Float16)(a0.z * SCALE); qa[dc][3]  = (_Float16)(a0.w * SCALE);
      qa[dc][4]  = (_Float16)(a1.x * SCALE); qa[dc][5]  = (_Float16)(a1.y * SCALE);
      qa[dc][6]  = (_Float16)(a1.z * SCALE); qa[dc][7]  = (_Float16)(a1.w * SCALE);
      qa[dc][8]  = (_Float16)(b0.x * SCALE); qa[dc][9]  = (_Float16)(b0.y * SCALE);
      qa[dc][10] = (_Float16)(b0.z * SCALE); qa[dc][11] = (_Float16)(b0.w * SCALE);
      qa[dc][12] = (_Float16)(b1.x * SCALE); qa[dc][13] = (_Float16)(b1.y * SCALE);
      qa[dc][14] = (_Float16)(b1.z * SCALE); qa[dc][15] = (_Float16)(b1.w * SCALE);
    }
  }

  v8f   o[4] = {};
  float rs[8] = {0.f,0.f,0.f,0.f,0.f,0.f,0.f,0.f};
  _Float16* wrow = &Wb[wave * 16 * LSTR];

  for (int kb = 0; kb < NKB; ++kb) {
    const int key0 = kb * BN;
    #pragma unroll
    for (int it = 0; it < 8; ++it) {
      int idx = it * 128 + tid;
      int r   = idx >> 4;
      int c4  = (idx & 15) << 2;
      float4 kk4 = *(const float4*)(K + (size_t)(key0 + r) * DHEAD + c4);
      float4 vv4 = *(const float4*)(V + (size_t)(key0 + r) * DHEAD + c4);
      v4h ksv;
      ksv[0] = (_Float16)kk4.x; ksv[1] = (_Float16)kk4.y;
      ksv[2] = (_Float16)kk4.z; ksv[3] = (_Float16)kk4.w;
      *(v4h*)&Ks[r * LSTR + c4] = ksv;
      Vt[(c4 + 0) * LSTR + r] = (_Float16)vv4.x;
      Vt[(c4 + 1) * LSTR + r] = (_Float16)vv4.y;
      Vt[(c4 + 2) * LSTR + r] = (_Float16)vv4.z;
      Vt[(c4 + 3) * LSTR + r] = (_Float16)vv4.w;
    }
    __syncthreads();

    #pragma unroll
    for (int n0t = 0; n0t < 4; ++n0t) {
      const int n0 = n0t * 16;
      v8f s = {};
      #pragma unroll
      for (int dc = 0; dc < 2; ++dc) {
        v16h bk = ld16h(&Ks[(n0 + l15) * LSTR + dc * 32 + 8 * hi]);
        s = __builtin_amdgcn_wmma_f32_16x16x32_f16(false, qa[dc], false, bk,
                                                   (short)0, s, false, false);
      }
      #pragma unroll
      for (int r = 0; r < 8; ++r) {
        float e = s[r];
        float w = e * e;
        rs[r] += w;
        wrow[(r + 8 * hi) * LSTR + n0 + l15] = (_Float16)w;
      }
    }
    asm volatile("s_wait_dscnt 0x0" ::: "memory");

    #pragma unroll
    for (int kc = 0; kc < 2; ++kc) {
      const _Float16* wp = &wrow[l15 * LSTR + kc * 32 + 8 * hi];
      v8h wlo = *(const v8h*)wp;
      v8h whi = *(const v8h*)(wp + 16);
      v16h aw = __builtin_shufflevector(wlo, whi,
                                        0,1,2,3,4,5,6,7,8,9,10,11,12,13,14,15);
      #pragma unroll
      for (int nt = 0; nt < 4; ++nt) {
        v16h bv = ld16h(&Vt[(nt * 16 + l15) * LSTR + kc * 32 + 8 * hi]);
        o[nt] = __builtin_amdgcn_wmma_f32_16x16x32_f16(false, aw, false, bv,
                                                       (short)0, o[nt], false, false);
      }
    }
    __syncthreads();
  }

  #pragma unroll
  for (int r = 0; r < 8; ++r) {
    float t = rs[r];
    t += __shfl_xor(t, 1, 32);
    t += __shfl_xor(t, 2, 32);
    t += __shfl_xor(t, 4, 32);
    t += __shfl_xor(t, 8, 32);
    const float inv = 1.0f / (t + EPSF);
    const int orow = row0 + r + 8 * hi;
    #pragma unroll
    for (int nt = 0; nt < 4; ++nt) {
      O[(size_t)orow * DHEAD + nt * 16 + l15] = o[nt][r] * inv;
    }
  }
}

extern "C" void kernel_launch(void* const* d_in, const int* in_sizes, int n_in,
                              void* d_out, int out_size, void* d_ws, size_t ws_size,
                              hipStream_t stream) {
  (void)in_sizes; (void)n_in; (void)out_size;
  const float* q = (const float*)d_in[0];
  const float* k = (const float*)d_in[1];
  const float* v = (const float*)d_in[2];
  float* out = (float*)d_out;

  const size_t elems = (size_t)BDIM * HDIM * NSEQ * DHEAD;   // 8.4M per tensor
  const size_t need  = 2 * elems * 2;                        // Kh + Vt, f16

  dim3 grid(NSEQ / BM, BDIM * HDIM);   // (32, 64)

  if (ws_size >= need) {
    _Float16* kh = (_Float16*)d_ws;
    _Float16* vt = kh + elems;
    convert_kv_kernel<<<grid, 128, 0, stream>>>(k, v, kh, vt);
    power_attn_p2_async_kernel<<<grid, 128, 0, stream>>>(q, kh, vt, out);
  } else {
    power_attn_p2_fallback<<<grid, 128, 0, stream>>>(q, k, v, out);
  }
}
